// ValueEstimator_60627758350778
// MI455X (gfx1250) — matrix operand order repaired and checked by reading, not compiled, not run on hardware
//
#include <hip/hip_runtime.h>

// Problem sizes (fixed by the reference)
#define Bsz   16384
#define INsz  512
#define HIDsz 2048
#define NE    8
#define NKT   (INsz / 32)    // 16  K-tiles of 32
#define NNT   (HIDsz / 16)   // 128 HID tiles of 16
#define NRT   (Bsz / 16)     // 1024 row tiles of 16
#define TILE_HALFS (NKT * 512)   // 8192 halfs = 16 KB per B tile

typedef __attribute__((ext_vector_type(16))) _Float16 v16h;
typedef __attribute__((ext_vector_type(8)))  float    v8f;
typedef __attribute__((ext_vector_type(4)))  int      v4i;

typedef __attribute__((address_space(1))) v4i gv4i;   // global b128
typedef __attribute__((address_space(3))) v4i lv4i;   // LDS b128

// ---------------------------------------------------------------------------
// gfx1250 async global->LDS copy (ASYNCcnt-tracked), builtin if available,
// inline-asm fallback otherwise. Builtin signature (probe-derived):
//   (v4i AS1* gsrc, v4i AS3* ldst, imm offset, imm cpol)
// ---------------------------------------------------------------------------
__device__ __forceinline__ void async_ld_b128(const void* g, void* l) {
#if __has_builtin(__builtin_amdgcn_global_load_async_to_lds_b128)
    __builtin_amdgcn_global_load_async_to_lds_b128((gv4i*)g, (lv4i*)l, 0, 0);
#else
    unsigned lofs = (unsigned)(size_t)(lv4i*)l;
    asm volatile("global_load_async_to_lds_b128 %0, %1, off"
                 :: "v"(lofs), "v"(g) : "memory");
#endif
}

__device__ __forceinline__ void wait_async0() {
#if __has_builtin(__builtin_amdgcn_s_wait_asynccnt)
    __builtin_amdgcn_s_wait_asynccnt(0);
#else
    asm volatile("s_wait_asynccnt 0x0" ::: "memory");
#endif
}

// ---------------------------------------------------------------------------
// Pack x (fp32 [B, IN]) into f16 WMMA A-fragments.
// A 16x32 layout: lanes 0-15 hold M=0..15 / K=0..15,
// lanes 16-31 hold M=0..15 / K=16..31; 16 contiguous halfs per lane.
// Fragment linear index = rt*NKT + kt.
// ---------------------------------------------------------------------------
__global__ void pack_x_kernel(const float* __restrict__ x, _Float16* __restrict__ xp) {
    const int gid  = blockIdx.x * blockDim.x + threadIdx.x;
    const int wave = gid >> 5;                 // = rt*NKT + kt
    const int lane = threadIdx.x & 31;
    const int kt   = wave & (NKT - 1);
    const int rt   = wave >> 4;
    if (rt >= NRT) return;
    const int row   = rt * 16 + (lane & 15);
    const int kbase = kt * 32 + ((lane >= 16) ? 16 : 0);
    const float* __restrict__ src = x + (size_t)row * INsz + kbase;
    v16h v;
#pragma unroll
    for (int h = 0; h < 16; ++h) {
        const int ko = h;  // 0..15 relative
        v[h] = (_Float16)src[ko];
    }
    *(v16h*)(xp + ((size_t)wave * 32 + lane) * 16) = v;
}

// ---------------------------------------------------------------------------
// Pack W1 (fp32 [E, IN, HID]) into f16 WMMA B-fragments (32x16 K x N).
// B layout: lanes 0-15 hold N=0..15 / K=0..15, lanes 16-31 hold N / K=16..31,
// 16 contiguous halfs (K ascending) per lane.
// Fragment linear index = (e*NNT + nt)*NKT + kt  -> one nt-tile is 16 KB
// contiguous, which the main kernel copies to LDS with async b128 loads.
// ---------------------------------------------------------------------------
__global__ void pack_w_kernel(const float* __restrict__ w1, _Float16* __restrict__ wp) {
    const int gid  = blockIdx.x * blockDim.x + threadIdx.x;
    const int wave = gid >> 5;                 // = (e*NNT + nt)*NKT + kt
    const int lane = threadIdx.x & 31;
    const int kt   = wave & (NKT - 1);
    const int nt   = (wave >> 4) & (NNT - 1);
    const int e    = wave >> 11;               // NKT*NNT = 2048 = 2^11
    if (e >= NE) return;
    const int n     = nt * 16 + (lane & 15);
    const int kbase = kt * 32 + ((lane >= 16) ? 16 : 0);
    const float* __restrict__ src = w1 + ((size_t)e * INsz + kbase) * HIDsz + n;
    v16h v;
#pragma unroll
    for (int h = 0; h < 16; ++h)
        v[h] = (_Float16)src[(size_t)h * HIDsz];
    *(v16h*)(wp + ((size_t)wave * 32 + lane) * 16) = v;
}

// ---------------------------------------------------------------------------
// Gating: one wave per row. logits = x[row] @ w_gate, top-4 + softmax,
// scattered back into a dense [B, E] gate matrix (zeros elsewhere).
// ---------------------------------------------------------------------------
__global__ void gate_kernel(const float* __restrict__ x, const float* __restrict__ wg,
                            float* __restrict__ gates) {
    const int gid  = blockIdx.x * blockDim.x + threadIdx.x;
    const int row  = gid >> 5;
    const int lane = threadIdx.x & 31;
    if (row >= Bsz) return;
    const float* __restrict__ xr = x + (size_t)row * INsz + lane * 16;
    float p[NE];
#pragma unroll
    for (int j = 0; j < NE; ++j) p[j] = 0.f;
#pragma unroll 4
    for (int t = 0; t < 16; ++t) {
        const float xv = xr[t];
        const float* __restrict__ wr = wg + (size_t)(lane * 16 + t) * NE;
#pragma unroll
        for (int j = 0; j < NE; ++j) p[j] = __builtin_fmaf(xv, wr[j], p[j]);
    }
#pragma unroll
    for (int off = 16; off >= 1; off >>= 1) {
#pragma unroll
        for (int j = 0; j < NE; ++j) p[j] += __shfl_xor(p[j], off, 32);
    }
    if (lane == 0) {
        bool sel[NE];
#pragma unroll
        for (int j = 0; j < NE; ++j) sel[j] = false;
        for (int r = 0; r < 4; ++r) {            // top-4 (first-index tie-break)
            int bi = 0; float bv = -3.4e38f;
            for (int j = 0; j < NE; ++j)
                if (!sel[j] && p[j] > bv) { bv = p[j]; bi = j; }
            sel[bi] = true;
        }
        float m = -3.4e38f;
        for (int j = 0; j < NE; ++j) if (sel[j] && p[j] > m) m = p[j];
        float s = 0.f, ev[NE];
        for (int j = 0; j < NE; ++j) { ev[j] = sel[j] ? expf(p[j] - m) : 0.f; s += ev[j]; }
        const float inv = 1.f / s;
        for (int j = 0; j < NE; ++j) gates[(size_t)row * NE + j] = ev[j] * inv;
    }
}

// ---------------------------------------------------------------------------
// Main fused MoE kernel. Block = 256 threads = 8 waves; wave w owns row tile
// rt = blockIdx.x*8 + w (16 rows) with all 16 A fragments in VGPRs.
// blockIdx.y picks 2 of the 8 experts. B tiles (16 KB, shared by all 8 waves)
// are double-buffered in LDS via async global->LDS b128 copies; WMMAs read
// fragments with ds_load_b128. Per hid tile: 16 chained WMMAs over K=512,
// fused bias+ReLU+(*W2 column) epilogue, cross-lane reduce per expert.
// ---------------------------------------------------------------------------
__global__ __launch_bounds__(256) void moe_wmma_kernel(
    const _Float16* __restrict__ xp, const _Float16* __restrict__ wp,
    const float* __restrict__ gates, const float* __restrict__ b1,
    const float* __restrict__ w2,    const float* __restrict__ b2,
    float* __restrict__ partial) {
    __shared__ __align__(16) _Float16 lbuf[2][TILE_HALFS];   // 2 x 16 KB

    const int tid     = threadIdx.x;
    const int lane    = tid & 31;
    const int wave    = tid >> 5;
    const int rt      = blockIdx.x * 8 + wave;
    const int rowbase = rt * 16;
    const int n       = lane & 15;
    const int rowadd  = (lane >= 16) ? 8 : 0;   // C/D: lanes 16-31 hold M=8..15
    const int eg      = blockIdx.y;             // experts eg*2, eg*2+1

    // A fragments for this wave's 16 rows: kept in VGPRs for the whole kernel.
    v16h a[NKT];
#pragma unroll
    for (int kt = 0; kt < NKT; ++kt)
        a[kt] = *(const v16h*)(xp + ((size_t)(rt * NKT + kt) * 32 + lane) * 16);

    float yacc[8];
#pragma unroll
    for (int j = 0; j < 8; ++j) yacc[j] = 0.f;

    // Cooperative async copy of one 16 KB B tile: 64 B (4 x b128) per thread.
    const _Float16* __restrict__ wbase = wp + (size_t)(eg * 2) * NNT * TILE_HALFS;
    const int coff = tid * 32;                   // halfs: 64 B per thread
    {   // prologue: tile 0 -> buffer 0
        const _Float16* g = wbase + coff;
        _Float16* l = &lbuf[0][coff];
#pragma unroll
        for (int q = 0; q < 4; ++q) async_ld_b128(g + q * 8, l + q * 8);
    }

    for (int ei = 0; ei < 2; ++ei) {
        const int e = eg * 2 + ei;
        const float* __restrict__ b1e = b1 + (size_t)e * HIDsz;
        const float* __restrict__ w2e = w2 + (size_t)e * HIDsz;
        float acc[8];
#pragma unroll
        for (int j = 0; j < 8; ++j) acc[j] = 0.f;

        for (int nt = 0; nt < NNT; ++nt) {
            const int t   = ei * NNT + nt;       // flat tile index 0..255
            const int sel = t & 1;

            wait_async0();                        // my slice of tile t landed
            __syncthreads();                      // everyone's slice landed

            const int tn = t + 1;                 // overlap copy of next tile
            if (tn < 2 * NNT) {
                const _Float16* g = wbase + (size_t)tn * TILE_HALFS + coff;
                _Float16* l = &lbuf[tn & 1][coff];
#pragma unroll
                for (int q = 0; q < 4; ++q) async_ld_b128(g + q * 8, l + q * 8);
            }

            const float b1v = b1e[nt * 16 + n];
            const float w2v = w2e[nt * 16 + n];
            v8f c = {};
#pragma unroll
            for (int kt = 0; kt < NKT; ++kt) {
                const v16h bfrag = *(const v16h*)&lbuf[sel][(kt * 32 + lane) * 16];
                c = __builtin_amdgcn_wmma_f32_16x16x32_f16(
                        /*neg_a=*/false, a[kt], /*neg_b=*/false, bfrag,
                        /*c_mod=*/(short)0, c, /*reuse_a=*/false, /*reuse_b=*/false);
            }
#pragma unroll
            for (int j = 0; j < 8; ++j) {         // bias + ReLU + *W2 column
                float v = c[j] + b1v;
                v = fmaxf(v, 0.f);
                acc[j] = __builtin_fmaf(v, w2v, acc[j]);
            }
            __syncthreads();                      // done reading lbuf[sel]
        }
#pragma unroll
        for (int j = 0; j < 8; ++j) {             // fold N across 16 lanes
            float s = acc[j];
            s += __shfl_xor(s, 1, 16);
            s += __shfl_xor(s, 2, 16);
            s += __shfl_xor(s, 4, 16);
            s += __shfl_xor(s, 8, 16);
            const int   row = rowbase + j + rowadd;
            const float g   = gates[(size_t)row * NE + e];
            yacc[j] += g * (s + b2[e]);
        }
    }
    if ((lane & 15) == 0) {
#pragma unroll
        for (int j = 0; j < 8; ++j)
            partial[(size_t)eg * Bsz + rowbase + j + rowadd] = yacc[j];
    }
}

// Deterministic fixed-order combine of the 4 expert-group partials.
__global__ void combine_kernel(const float* __restrict__ partial, float* __restrict__ out) {
    const int b = blockIdx.x * blockDim.x + threadIdx.x;
    if (b >= Bsz) return;
    out[b] = ((partial[b] + partial[(size_t)Bsz + b]) +
              partial[(size_t)2 * Bsz + b]) + partial[(size_t)3 * Bsz + b];
}

// ---------------------------------------------------------------------------
extern "C" void kernel_launch(void* const* d_in, const int* in_sizes, int n_in,
                              void* d_out, int out_size, void* d_ws, size_t ws_size,
                              hipStream_t stream) {
    const float* x  = (const float*)d_in[0];   // [B, IN]
    const float* wg = (const float*)d_in[1];   // [IN, E]
    const float* W1 = (const float*)d_in[2];   // [E, IN, HID]
    const float* b1 = (const float*)d_in[3];   // [E, HID]
    const float* W2 = (const float*)d_in[4];   // [E, HID, 1]
    const float* b2 = (const float*)d_in[5];   // [E, 1]
    float* out = (float*)d_out;                // [B, 1] fp32

    // Workspace layout (~34.3 MB total)
    char* ws = (char*)d_ws;
    float*    gates   = (float*)ws;                                  // B*E*4   = 512 KB
    float*    partial = (float*)(ws + (size_t)Bsz * NE * 4);         // 4*B*4   = 256 KB
    _Float16* xp      = (_Float16*)(ws + (size_t)Bsz * NE * 4 + (size_t)4 * Bsz * 4);
    _Float16* wp      = xp + (size_t)Bsz * INsz;                     // each 16.8 MB

    pack_x_kernel<<<dim3((NRT * NKT * 32) / 256), 256, 0, stream>>>(x, xp);
    pack_w_kernel<<<dim3((NE * NNT * NKT * 32) / 256), 256, 0, stream>>>(W1, wp);
    gate_kernel<<<dim3((Bsz * 32) / 256), 256, 0, stream>>>(x, wg, gates);
    moe_wmma_kernel<<<dim3(Bsz / 128, 4), 256, 0, stream>>>(xp, wp, gates, b1, W2, b2, partial);
    combine_kernel<<<dim3(Bsz / 256), 256, 0, stream>>>(partial, out);

    (void)in_sizes; (void)n_in; (void)out_size; (void)ws_size;
}
